// EquivariantNetwork_60902636257739
// MI455X (gfx1250) — matrix operand order repaired and checked by reading, not compiled, not run on hardware
//
#include <hip/hip_runtime.h>
#include <hip/hip_bf16.h>
#include <math.h>

#define NN 20000
#define NE 320000
#define NTY 4
#define NS 128
#define NV 32
#define NRAD 8
#define MLPD 64
#define WDIM 320   // 2*NS + 2*NV
#define NT 10
#define RBLK 256

typedef __attribute__((ext_vector_type(16))) _Float16 v16h;
typedef __attribute__((ext_vector_type(8)))  _Float16 v8h;
typedef __attribute__((ext_vector_type(8)))  float    v8f;

union frag16 { v16h v; v8h h[2]; };

__device__ inline v8f wmma_f16(v16h a, v16h b, v8f c) {
    return __builtin_amdgcn_wmma_f32_16x16x32_f16(false, a, false, b, (short)0, c, false, false);
}

// A fragment (16x32 f16), row-major f16 source, ld in elements (multiple of 8).
// lane: row = lane&15, elements t<8 -> k=8*lh+t, t>=8 -> k=8*lh+16+(t-8): two 16B loads.
__device__ inline v16h load_a16(const _Float16* __restrict__ src, int ld, int lane) {
    int row = lane & 15, lh = lane >> 4;
    const _Float16* p = src + row * ld + 8 * lh;
    frag16 u;
    u.h[0] = *(const v8h*)(p);
    u.h[1] = *(const v8h*)(p + 16);
    return u.v;
}
// B fragment (32x16 f16) from TRANSPOSED weights: srcT[c][k], ldk elements.
// lane: col = lane&15, t<8 -> k=8*lh+t, t>=8 -> k=8*lh+16+(t-8): two 16B loads.
__device__ inline v16h load_b16T(const _Float16* __restrict__ srcT, int ldk, int colBase,
                                 int kBase, int lane) {
    int col = lane & 15, lh = lane >> 4;
    const _Float16* p = srcT + (colBase + col) * ldk + kBase + 8 * lh;
    frag16 u;
    u.h[0] = *(const v8h*)(p);
    u.h[1] = *(const v8h*)(p + 16);
    return u.v;
}

// ---------------- geometry: r, y1, bessel*cutoff radial basis (f16) ----------------
__global__ void geom_kernel(const float* __restrict__ ev, float* __restrict__ y1g,
                            _Float16* __restrict__ rad_h) {
    int e = blockIdx.x * blockDim.x + threadIdx.x;
    if (e >= NE) return;
    float x = ev[e * 3 + 0], y = ev[e * 3 + 1], z = ev[e * 3 + 2];
    float r = sqrtf(x * x + y * y + z * z + 1e-12f);
    float rinv = 1.0f / (r + 1e-9f);
    y1g[e * 3 + 0] = x * rinv;
    y1g[e * 3 + 1] = y * rinv;
    y1g[e * 3 + 2] = z * rinv;
    float u = r * 0.2f;                     // r / R_MAX
    float u2 = u * u, u6 = u2 * u2 * u2;
    float fcut = 1.0f - 28.0f * u6 + 48.0f * u6 * u - 21.0f * u6 * u2;
    if (u >= 1.0f) fcut = 0.0f;
    float pref = sqrtf(0.4f) * rinv * fcut; // sqrt(2/R_MAX)
    const float pir = 3.14159265358979323846f * r * 0.2f;
#pragma unroll
    for (int k = 0; k < NRAD; ++k)
        rad_h[e * NRAD + k] = (_Float16)(pref * sinf((float)(k + 1) * pir));
}

// ---------------- utility kernels ----------------
__global__ void zero_kernel(float* __restrict__ p, int n) {
    for (int i = blockIdx.x * blockDim.x + threadIdx.x; i < n; i += gridDim.x * blockDim.x)
        p[i] = 0.0f;
}
// transpose+convert: dst[c*Kpad + k] = (k<K) ? src[(rowOff+k)*N + c] : 0   (f32 -> f16)
__global__ void transpose_f16_kernel(const float* __restrict__ src, _Float16* __restrict__ dst,
                                     int K, int N, int Kpad, int rowOff) {
    int i = blockIdx.x * blockDim.x + threadIdx.x;
    if (i >= N * Kpad) return;
    int c = i / Kpad, k = i % Kpad;
    dst[i] = (k < K) ? (_Float16)src[(rowOff + k) * N + c] : (_Float16)0.0f;
}
// layer 0: Ts[n] = Wss0[type[n]], Tsv[n] = Wsv0[type[n]]  (onehot @ W == row select)
__global__ void fill_ts0_kernel(const int* __restrict__ types, const float* __restrict__ Wss0,
                                const float* __restrict__ Wsv0, float* __restrict__ Ts,
                                float* __restrict__ Tsv) {
    int i = blockIdx.x * blockDim.x + threadIdx.x;
    if (i < NN * NS) { int n = i / NS, c = i % NS; Ts[i] = Wss0[types[n] * NS + c]; }
    if (i < NN * NV) { int n = i / NV, c = i % NV; Tsv[i] = Wsv0[types[n] * NV + c]; }
}

// ---------------- per-node hoisted transforms (WMMA f16, vectorized fragments) ----------------
// Tasks: 40 per 16-row tile: 8 Ts ntiles, 2 Tsv, 24 Vp (3 dims x 8), 6 Tv (3 x 2)
__global__ void __launch_bounds__(256) node_transform_kernel(
    const _Float16* __restrict__ s_h, const _Float16* __restrict__ v_h,
    const int* __restrict__ types,
    const _Float16* __restrict__ WssT, const _Float16* __restrict__ WsvT,
    const _Float16* __restrict__ WvsT, const _Float16* __restrict__ WvvT,
    const float* __restrict__ Wss, const float* __restrict__ Wsv,
    float* __restrict__ Ts, float* __restrict__ Tsv,
    float* __restrict__ Vp, float* __restrict__ Tv) {
    int wave = (blockIdx.x * blockDim.x + threadIdx.x) >> 5;
    int lane = threadIdx.x & 31;
    int rt = wave / 40, ot = wave % 40;
    if (rt >= NN / 16) return;                 // wave-uniform guard
    int n0 = rt * 16;
    int col = lane & 15, lh = lane >> 4;
    v8f acc = {};
    if (ot < 8) {                              // Ts = s @ Wss[4:,:]  (K=128)
        int nt = ot;
        for (int ks = 0; ks < 4; ++ks) {
            v16h a = load_a16(s_h + n0 * NS + ks * 32, NS, lane);
            v16h b = load_b16T(WssT, NS, nt * 16, ks * 32, lane);
            acc = wmma_f16(a, b, acc);
        }
        int cg = nt * 16 + col;
#pragma unroll
        for (int r = 0; r < 8; ++r) {
            int n = n0 + r + 8 * lh;
            Ts[n * NS + cg] = acc[r] + Wss[types[n] * NS + cg]; // + onehot rows
        }
    } else if (ot < 10) {                      // Tsv = s @ Wsv[4:,:]  (K=128)
        int nt = ot - 8;
        for (int ks = 0; ks < 4; ++ks) {
            v16h a = load_a16(s_h + n0 * NS + ks * 32, NS, lane);
            v16h b = load_b16T(WsvT, NS, nt * 16, ks * 32, lane);
            acc = wmma_f16(a, b, acc);
        }
        int cg = nt * 16 + col;
#pragma unroll
        for (int r = 0; r < 8; ++r) {
            int n = n0 + r + 8 * lh;
            Tsv[n * NV + cg] = acc[r] + Wsv[types[n] * NV + cg];
        }
    } else if (ot < 34) {                      // Vp[:,i,:] = v[:,:,i] @ Wvs  (K=32)
        int i = (ot - 10) / 8, nt = (ot - 10) % 8;
        v16h a = load_a16(v_h + n0 * 96 + i * 32, 96, lane);
        v16h b = load_b16T(WvsT, NV, nt * 16, 0, lane);
        acc = wmma_f16(a, b, acc);
        int cg = nt * 16 + col;
#pragma unroll
        for (int r = 0; r < 8; ++r) {
            int n = n0 + r + 8 * lh;
            Vp[n * 384 + i * 128 + cg] = acc[r];
        }
    } else {                                   // Tv[:,i,:] = v[:,:,i] @ Wvv  (K=32)
        int i = (ot - 34) / 2, nt = (ot - 34) % 2;
        v16h a = load_a16(v_h + n0 * 96 + i * 32, 96, lane);
        v16h b = load_b16T(WvvT, NV, nt * 16, 0, lane);
        acc = wmma_f16(a, b, acc);
        int cg = nt * 16 + col;
#pragma unroll
        for (int r = 0; r < 8; ++r) {
            int n = n0 + r + 8 * lh;
            Tv[n * 96 + i * 32 + cg] = acc[r];
        }
    }
}

// ---------------- fused radial MLP (WMMA) + message assembly + atomic scatter ----------------
// block = 128 threads = 4 waves; each wave owns 16 edges; LDS: h 2KB/wave, w 10KB/wave (48KB)
__global__ void __launch_bounds__(128) edge_message_kernel(
    const _Float16* __restrict__ rad_h, const float* __restrict__ y1g,
    const int* __restrict__ src, const int* __restrict__ dst,
    const _Float16* __restrict__ W1T, const _Float16* __restrict__ W2T,
    const float* __restrict__ b1,
    const float* __restrict__ Ts, const float* __restrict__ Tsv,
    const float* __restrict__ Vp, const float* __restrict__ Tv,
    float* __restrict__ s_msg, float* __restrict__ v_msg, int has_v) {
    __shared__ _Float16 h_lds[4][16][MLPD];
    __shared__ _Float16 w_lds[4][16][WDIM];
    int wid = threadIdx.x >> 5;
    int lane = threadIdx.x & 31;
    int e0 = blockIdx.x * 64 + wid * 16;
    int row = lane & 15, lh = lane >> 4, col = lane & 15;

    // phase 1: h = silu(rad @ W1 + b1)   (K=8 zero-padded to 32)
    frag16 ua;
    v8h zz = {};
    ua.h[0] = (lh == 0) ? *(const v8h*)(rad_h + (e0 + row) * NRAD) : zz; // k=0..7 real
    ua.h[1] = zz;                                                        // k>=16 pad
    v16h a1 = ua.v;
    for (int nt = 0; nt < 4; ++nt) {
        v16h b = load_b16T(W1T, 32, nt * 16, 0, lane);  // W1T[64][32] (K padded)
        v8f acc = {};
        acc = wmma_f16(a1, b, acc);
        float bb = b1[nt * 16 + col];
#pragma unroll
        for (int r = 0; r < 8; ++r) {
            float x = acc[r] + bb;
            h_lds[wid][r + 8 * lh][nt * 16 + col] = (_Float16)(x / (1.0f + __expf(-x)));
        }
    }
    // phase 2: w = h @ W2   (K=64, N=320), A from LDS (ds_load_b128), B from W2T
    for (int nt = 0; nt < 20; ++nt) {
        v8f acc = {};
        for (int ks = 0; ks < 2; ++ks) {
            v16h a = load_a16(&h_lds[wid][0][0] + ks * 32, MLPD, lane);
            v16h b = load_b16T(W2T, MLPD, nt * 16, ks * 32, lane);
            acc = wmma_f16(a, b, acc);
        }
#pragma unroll
        for (int r = 0; r < 8; ++r)
            w_lds[wid][r + 8 * lh][nt * 16 + col] = (_Float16)acc[r];
    }
    // phase 3: messages; gathers hit L2-resident node tables; L2 atomic scatter
    const float inv = 0.25f;  // 1/sqrt(E/N)
    for (int m = 0; m < 16; ++m) {
        int e = e0 + m;
        int j = src[e], d = dst[e];
        float yx = y1g[e * 3 + 0], yy = y1g[e * 3 + 1], yz = y1g[e * 3 + 2];
#pragma unroll
        for (int c0 = 0; c0 < 4; ++c0) {
            int c = c0 * 32 + lane;
            float msg = (float)w_lds[wid][m][c] * Ts[j * NS + c];
            if (has_v) {
                float dp = yx * Vp[j * 384 + c] + yy * Vp[j * 384 + 128 + c] +
                           yz * Vp[j * 384 + 256 + c];
                msg += (float)w_lds[wid][m][NS + c] * dp;
            }
            unsafeAtomicAdd(&s_msg[d * NS + c], msg * inv);
        }
        int o = lane;
        float sv = (float)w_lds[wid][m][2 * NS + o] * Tsv[j * NV + o];
        float wvv = has_v ? (float)w_lds[wid][m][2 * NS + NV + o] : 0.0f;
        float yv[3] = {yx, yy, yz};
#pragma unroll
        for (int i = 0; i < 3; ++i) {
            float msg = sv * yv[i];
            if (has_v) msg += wvv * Tv[j * 96 + i * 32 + o];
            unsafeAtomicAdd(&v_msg[d * 96 + o * 3 + i], msg * inv);
        }
    }
}

// ---------------- equivariant batchnorm ----------------
__global__ void bn_stats_kernel(const float* __restrict__ s_msg, const float* __restrict__ v_msg,
                                float* __restrict__ stats) {
    int c = threadIdx.x;                        // blockDim = 160
    int per = (NN + gridDim.x - 1) / gridDim.x;
    int nlo = blockIdx.x * per;
    int nhi = nlo + per; if (nhi > NN) nhi = NN;
    if (c < NS) {
        float s1 = 0.0f, s2 = 0.0f;
        for (int n = nlo; n < nhi; ++n) { float v = s_msg[n * NS + c]; s1 += v; s2 += v * v; }
        atomicAdd(&stats[c], s1);
        atomicAdd(&stats[NS + c], s2);
    } else if (c < NS + NV) {
        int o = c - NS;
        float s = 0.0f;
        for (int n = nlo; n < nhi; ++n) {
            float a = v_msg[n * 96 + o * 3 + 0], b = v_msg[n * 96 + o * 3 + 1],
                  d = v_msg[n * 96 + o * 3 + 2];
            s += a * a + b * b + d * d;
        }
        atomicAdd(&stats[2 * NS + o], s);
    }
}
// also maintains f16 mirrors s_h [N][128], v_h [N][3][32] for WMMA A-operands
__global__ void bn_apply_kernel(const float* __restrict__ s_msg, const float* __restrict__ v_msg,
                                const float* __restrict__ stats,
                                const float* __restrict__ bn_ws, const float* __restrict__ bn_bs,
                                const float* __restrict__ bn_wv,
                                float* __restrict__ s_cur, float* __restrict__ v_cur,
                                _Float16* __restrict__ s_h, _Float16* __restrict__ v_h,
                                int first) {
    int idx = blockIdx.x * blockDim.x + threadIdx.x;
    if (idx >= NN * 160) return;
    int n = idx / 160, ch = idx % 160;
    const float invN = 1.0f / (float)NN;
    if (ch < NS) {
        float mean = stats[ch] * invN;
        float var = stats[NS + ch] * invN - mean * mean;
        float us = (s_msg[n * NS + ch] - mean) * rsqrtf(var + 1e-5f) * bn_ws[ch] + bn_bs[ch];
        float sn = first ? us : (s_cur[n * NS + ch] + us);
        s_cur[n * NS + ch] = sn;
        s_h[n * NS + ch] = (_Float16)sn;
    } else {
        int o = ch - NS;
        float scale = rsqrtf(stats[2 * NS + o] * invN + 1e-5f) * bn_wv[o];
#pragma unroll
        for (int i = 0; i < 3; ++i) {
            int id = n * 96 + o * 3 + i;
            float uv = v_msg[id] * scale;
            float vn = first ? uv : (v_cur[id] + uv);
            v_cur[id] = vn;
            v_h[n * 96 + i * 32 + o] = (_Float16)vn;   // i-major layout for A-fragments
        }
    }
}

// ---------------- stable sort by type + readout ----------------
__global__ void count_types_kernel(const int* __restrict__ types, int* __restrict__ bc) {
    __shared__ int cnt[NTY];
    if (threadIdx.x < NTY) cnt[threadIdx.x] = 0;
    __syncthreads();
    int n = blockIdx.x * RBLK + threadIdx.x;
    if (n < NN) atomicAdd(&cnt[types[n]], 1);
    __syncthreads();
    if (threadIdx.x < NTY) bc[blockIdx.x * NTY + threadIdx.x] = cnt[threadIdx.x];
}
__global__ void scan_offsets_kernel(const int* __restrict__ bc, int* __restrict__ bo,
                                    int* __restrict__ base, int nblocks) {
    if (blockIdx.x == 0 && threadIdx.x == 0) {
        int tot[NTY] = {0, 0, 0, 0};
        for (int b = 0; b < nblocks; ++b)
            for (int t = 0; t < NTY; ++t) { bo[b * NTY + t] = tot[t]; tot[t] += bc[b * NTY + t]; }
        int acc = 0;
        for (int t = 0; t < NTY; ++t) { base[t] = acc; acc += tot[t]; }
    }
}
__global__ void readout_kernel(const int* __restrict__ types, const float* __restrict__ s_cur,
                               const float* __restrict__ v_cur, const float* __restrict__ Wr,
                               const float* __restrict__ br, const int* __restrict__ bo,
                               const int* __restrict__ base, float* __restrict__ out) {
    __shared__ int tsh[RBLK];
    int n = blockIdx.x * RBLK + threadIdx.x;
    int t = (n < NN) ? types[n] : -1;
    tsh[threadIdx.x] = t;
    __syncthreads();
    if (n >= NN) return;
    int rank = 0;
    for (int j = 0; j < (int)threadIdx.x; ++j) rank += (tsh[j] == t);
    int dest = base[t] + bo[blockIdx.x * NTY + t] + rank;
    float acc[NT];
#pragma unroll
    for (int o = 0; o < NT; ++o) acc[o] = br[t * NT + o];
    for (int f = 0; f < NS; ++f) {
        float fv = s_cur[n * NS + f];
#pragma unroll
        for (int o = 0; o < NT; ++o) acc[o] += fv * Wr[(t * 160 + f) * NT + o];
    }
    for (int c = 0; c < NV; ++c) {
        float a = v_cur[n * 96 + c * 3 + 0], b = v_cur[n * 96 + c * 3 + 1],
              d = v_cur[n * 96 + c * 3 + 2];
        float fv = sqrtf(a * a + b * b + d * d + 1e-12f);
#pragma unroll
        for (int o = 0; o < NT; ++o) acc[o] += fv * Wr[(t * 160 + NS + c) * NT + o];
    }
#pragma unroll
    for (int o = 0; o < NT; ++o) out[dest * NT + o] = acc[o];
}

// ---------------- host orchestration ----------------
extern "C" void kernel_launch(void* const* d_in, const int* in_sizes, int n_in,
                              void* d_out, int out_size, void* d_ws, size_t ws_size,
                              hipStream_t stream) {
    // Input order = jax pytree flatten (sorted dict keys):
    // 0 node_types, 1 edge_index[2,E], 2 edge_vec, 3 Wr, 4 br,
    // then per layer l (base 5+l*10): W1, W2, Wss, Wsv, Wvs, Wvv, b1, bn_bs, bn_ws, bn_wv
    const int*   node_types = (const int*)d_in[0];
    const int*   ei         = (const int*)d_in[1];
    const float* edge_vec   = (const float*)d_in[2];
    const float* Wr         = (const float*)d_in[3];
    const float* br         = (const float*)d_in[4];
    const int* src = ei;
    const int* dstp = ei + NE;

    char* wsb = (char*)d_ws;
    size_t off = 0;
    auto alloc = [&](size_t bytes) -> void* {
        off = (off + 255) & ~(size_t)255;
        void* p = wsb + off;
        off += bytes;
        return p;
    };
    float*    y1    = (float*)alloc((size_t)NE * 3 * 4);
    _Float16* rad_h = (_Float16*)alloc((size_t)NE * NRAD * 2);
    float*    s_cur = (float*)alloc((size_t)NN * NS * 4);
    float*    v_cur = (float*)alloc((size_t)NN * 96 * 4);
    _Float16* s_h   = (_Float16*)alloc((size_t)NN * NS * 2);
    _Float16* v_h   = (_Float16*)alloc((size_t)NN * 96 * 2);
    float*    s_msg = (float*)alloc((size_t)NN * NS * 4);
    float*    v_msg = (float*)alloc((size_t)NN * 96 * 4);
    float*    Ts    = (float*)alloc((size_t)NN * NS * 4);
    float*    Tsv   = (float*)alloc((size_t)NN * NV * 4);
    float*    Vp    = (float*)alloc((size_t)NN * 384 * 4);
    float*    Tv    = (float*)alloc((size_t)NN * 96 * 4);
    _Float16* WssT  = (_Float16*)alloc(128 * 128 * 2);  // [c][k] k=0..127 (rows 4..131)
    _Float16* WsvT  = (_Float16*)alloc(32 * 128 * 2);
    _Float16* WvsT  = (_Float16*)alloc(128 * 32 * 2);
    _Float16* WvvT  = (_Float16*)alloc(32 * 32 * 2);
    _Float16* W1T   = (_Float16*)alloc(64 * 32 * 2);    // [c][k] K padded 8->32
    _Float16* W2T   = (_Float16*)alloc(320 * 64 * 2);
    float*    stats = (float*)alloc(288 * 4);
    const int nrb = (NN + RBLK - 1) / RBLK;  // 79
    int* blkCnt = (int*)alloc((size_t)nrb * NTY * 4);
    int* blkOff = (int*)alloc((size_t)nrb * NTY * 4);
    int* tybase = (int*)alloc(NTY * 4);

    geom_kernel<<<NE / 256, 256, 0, stream>>>(edge_vec, y1, rad_h);

    for (int l = 0; l <= 4; ++l) {
        const float* W1   = (const float*)d_in[5 + l * 10 + 0];
        const float* W2   = (const float*)d_in[5 + l * 10 + 1];
        const float* Wss  = (const float*)d_in[5 + l * 10 + 2];
        const float* Wsv  = (const float*)d_in[5 + l * 10 + 3];
        const float* Wvs  = (const float*)d_in[5 + l * 10 + 4];
        const float* Wvv  = (const float*)d_in[5 + l * 10 + 5];
        const float* b1   = (const float*)d_in[5 + l * 10 + 6];
        const float* bnbs = (const float*)d_in[5 + l * 10 + 7];
        const float* bnws = (const float*)d_in[5 + l * 10 + 8];
        const float* bnwv = (const float*)d_in[5 + l * 10 + 9];

        transpose_f16_kernel<<<8, 256, 0, stream>>>(W1, W1T, NRAD, MLPD, 32, 0);
        transpose_f16_kernel<<<80, 256, 0, stream>>>(W2, W2T, MLPD, WDIM, MLPD, 0);

        if (l == 0) {
            fill_ts0_kernel<<<(NN * NS + 255) / 256, 256, 0, stream>>>(node_types, Wss, Wsv, Ts, Tsv);
        } else {
            transpose_f16_kernel<<<64, 256, 0, stream>>>(Wss, WssT, 128, NS, 128, 4);
            transpose_f16_kernel<<<16, 256, 0, stream>>>(Wsv, WsvT, 128, NV, 128, 4);
            transpose_f16_kernel<<<16, 256, 0, stream>>>(Wvs, WvsT, 32, NS, 32, 0);
            transpose_f16_kernel<<<4, 256, 0, stream>>>(Wvv, WvvT, 32, NV, 32, 0);
            // (NN/16)*40 wave-tasks / 8 waves per block
            node_transform_kernel<<<(NN / 16) * 40 / 8, 256, 0, stream>>>(
                s_h, v_h, node_types, WssT, WsvT, WvsT, WvvT, Wss, Wsv,
                Ts, Tsv, Vp, Tv);
        }

        zero_kernel<<<2048, 256, 0, stream>>>(s_msg, NN * NS);
        zero_kernel<<<2048, 256, 0, stream>>>(v_msg, NN * 96);

        edge_message_kernel<<<NE / 64, 128, 0, stream>>>(
            rad_h, y1, src, dstp, W1T, W2T, b1, Ts, Tsv, Vp, Tv,
            s_msg, v_msg, (l > 0) ? 1 : 0);

        zero_kernel<<<2, 256, 0, stream>>>(stats, 288);
        bn_stats_kernel<<<100, 160, 0, stream>>>(s_msg, v_msg, stats);
        bn_apply_kernel<<<(NN * 160 + 255) / 256, 256, 0, stream>>>(
            s_msg, v_msg, stats, bnws, bnbs, bnwv, s_cur, v_cur, s_h, v_h, (l == 0) ? 1 : 0);
    }

    count_types_kernel<<<nrb, RBLK, 0, stream>>>(node_types, blkCnt);
    scan_offsets_kernel<<<1, 32, 0, stream>>>(blkCnt, blkOff, tybase, nrb);
    readout_kernel<<<nrb, RBLK, 0, stream>>>(node_types, s_cur, v_cur, Wr, br,
                                             blkOff, tybase, (float*)d_out);
    (void)in_sizes; (void)n_in; (void)out_size; (void)ws_size;
}
